// FNOFactorizedBlock2D_42984032698427
// MI455X (gfx1250) — matrix operand order repaired and checked by reading, not compiled, not run on hardware
//
#include <hip/hip_runtime.h>
#include <math.h>

// ---------------- CDNA5 WMMA plumbing (wave32, 16x16x32 bf16, f32 accum) ----
typedef __bf16 bf16;
typedef __attribute__((ext_vector_type(16))) __bf16 v16bf;
typedef __attribute__((ext_vector_type(8)))  float  v8f;

union V16U { v16bf v; uint4 q[2]; };

__device__ __forceinline__ bf16 tobf(float f) { return (bf16)f; }

// A tile 16x32 (MxK), row-major source, ld in elements (multiple of 8).
// Lane layout (ISA 7.12.2): lanes 0-15 hold M=lane, K in {0..7,16..23};
// lanes 16-31 hold M=lane-16, K in {8..15,24..31}. Two b128 loads per lane.
__device__ __forceinline__ v16bf load_a16(const bf16* tile, int lda) {
  const int lane = threadIdx.x & 31;
  const int m  = lane & 15;
  const int kb = (lane >> 4) << 3;     // 0 or 8
  const bf16* p = tile + m * lda + kb;
  V16U u;
  u.q[0] = *(const uint4*)(p);
  u.q[1] = *(const uint4*)(p + 16);
  return u.v;
}

// B tile 32x16 (KxN) given as B-transposed (NxK row-major, K contiguous).
// lanes 0-15: K {0..7,16..23} of column N=lane; lanes 16-31: K {8..15,24..31} of N=lane-16.
__device__ __forceinline__ v16bf load_bt16(const bf16* bt, int ldb) {
  const int lane = threadIdx.x & 31;
  const int n  = lane & 15;
  const int kb = (lane >> 4) << 3;     // 0 or 8
  const bf16* p = bt + n * ldb + kb;
  V16U u;
  u.q[0] = *(const uint4*)(p);
  u.q[1] = *(const uint4*)(p + 16);
  return u.v;
}

__device__ __forceinline__ v8f wmma_bf16(v16bf a, v16bf b, v8f c) {
  // args: (neg_a, A, neg_b, B, c_mod, C, reuse_a, reuse_b)
  return __builtin_amdgcn_wmma_f32_16x16x32_bf16(false, a, false, b, (short)0, c,
                                                 false, false);
}

// C/D 16x16 f32: element r of v8f -> row M = r + (lane>=16 ? 8 : 0), col N = lane&15.
__device__ __forceinline__ v8f bias_acc(float bias) {
  v8f acc;
#pragma unroll
  for (int r = 0; r < 8; ++r) acc[r] = bias;
  return acc;
}

// workspace layout (bf16 element offsets, all 16B aligned)
#define WS_INW   0L          // 64x32  input weight, K padded 12->32
#define WS_BCW1  2048L       // 4 x 128x64
#define WS_BCW2  34816L      // 4 x 64x128
#define WS_OW1   67584L      // 128x64
#define WS_OW2   75776L      // 16x128 (out padded 5->16)
#define WS_RW1   77824L      // 128x64
#define WS_RW2   86016L      // 64x128
#define WS_DFT   94208L      // 64x256  rows 0-31 Re, 32-63 Im of kept bins
#define WS_IDFT  110592L     // 256x64  cols 0-31 from Gr, 32-63 from Gi
#define WS_MWR   126976L     // 4l x 2dir x 32k x (64o x 64i)  Re(w)
#define WS_MWI   1175552L    //                                Im(w)
#define WS_MWN   2224128L    //                               -Im(w)
#define WS_FR    3272704L    // 32k x 2048s x 64i  Re(F) bf16
#define WS_FI    7467008L    // 32k x 2048s x 64i  Im(F) bf16
#define WS_GT    11661312L   // 64kcat x 2048s x 64o  (Gr k=0..31 | Gi k=0..31)
// total 20,049,920 bf16 = ~40.1 MB

// ---------------- prep: bf16 weights, DFT bases, mode coefficients ----------
__global__ void fno_prep(const float* __restrict__ in_w,
                         const float* __restrict__ bc_w1,
                         const float* __restrict__ bc_w2,
                         const float* __restrict__ out_w1,
                         const float* __restrict__ out_w2,
                         const float* __restrict__ res_w1,
                         const float* __restrict__ res_w2,
                         const float* __restrict__ fw,
                         bf16* __restrict__ ws) {
  const float PI = 3.14159265358979323846f;
  const long total = 1175552L;
  for (long idx = (long)blockIdx.x * blockDim.x + threadIdx.x; idx < total;
       idx += (long)gridDim.x * blockDim.x) {
    long i = idx;
    if (i < 2048) {            // input weight padded
      int o = (int)(i >> 5), k = (int)(i & 31);
      ws[WS_INW + i] = (k < 12) ? tobf(in_w[o * 12 + k]) : tobf(0.f);
      continue;
    }
    i -= 2048;
    if (i < 32768) { ws[WS_BCW1 + i] = tobf(bc_w1[i]); continue; }
    i -= 32768;
    if (i < 32768) { ws[WS_BCW2 + i] = tobf(bc_w2[i]); continue; }
    i -= 32768;
    if (i < 8192)  { ws[WS_OW1 + i] = tobf(out_w1[i]); continue; }
    i -= 8192;
    if (i < 2048) {            // out_w2 padded 5->16 rows
      int o = (int)(i >> 7), k = (int)(i & 127);
      ws[WS_OW2 + i] = (o < 5) ? tobf(out_w2[o * 128 + k]) : tobf(0.f);
      continue;
    }
    i -= 2048;
    if (i < 8192)  { ws[WS_RW1 + i] = tobf(res_w1[i]); continue; }
    i -= 8192;
    if (i < 8192)  { ws[WS_RW2 + i] = tobf(res_w2[i]); continue; }
    i -= 8192;
    if (i < 16384) {           // forward DFT rows (ortho rfft at kept bins)
      int r = (int)(i >> 8), n = (int)(i & 255);
      int mode = r & 31;
      int bin = (mode < 16) ? mode : mode + 97;   // 113..128 for high modes
      float th = 2.f * PI * (float)bin * (float)n * (1.f / 256.f);
      float v = (r < 32) ? (cosf(th) * (1.f / 16.f)) : (-sinf(th) * (1.f / 16.f));
      ws[WS_DFT + i] = tobf(v);
      continue;
    }
    i -= 16384;
    if (i < 16384) {           // inverse DFT (ortho irfft from kept bins)
      int n = (int)(i >> 6), k = (int)(i & 63);
      int mode = k & 31;
      int bin = (mode < 16) ? mode : mode + 97;
      int edge = (bin == 0 || bin == 128);
      float a = edge ? 1.f : 2.f;
      float th = 2.f * PI * (float)bin * (float)n * (1.f / 256.f);
      float v;
      if (k < 32) v = a * cosf(th) * (1.f / 16.f);               // weight on Re
      else        v = edge ? 0.f : -a * sinf(th) * (1.f / 16.f); // weight on Im
      ws[WS_IDFT + i] = tobf(v);
      continue;
    }
    i -= 16384;
    {                          // spectral mixing weights, transposed [k][o][i]
      int ii  = (int)(i & 63);
      int o   = (int)((i >> 6) & 63);
      int k   = (int)((i >> 12) & 31);
      int dir = (int)((i >> 17) & 1);
      int l   = (int)(i >> 18);
      int d = (dir == 0) ? (k < 16 ? 0 : 2) : (k < 16 ? 1 : 3);
      int y = k & 15;
      long sb = ((((long)(l * 4 + d) * 64 + ii) * 64 + o) * 16 + y) * 2;
      float wr = fw[sb], wi = fw[sb + 1];
      long dst = ((long)((l * 2 + dir) * 32 + k)) * 4096 + (long)o * 64 + ii;
      ws[WS_MWR + dst] = tobf(wr);
      ws[WS_MWI + dst] = tobf(wi);
      ws[WS_MWN + dst] = tobf(-wi);
    }
  }
}

// ---------------- input projection: h = x @ in_w^T + in_b -------------------
__global__ void fno_in(const float* __restrict__ x, const bf16* __restrict__ ws,
                       const float* __restrict__ in_b, float* __restrict__ h) {
  __shared__ alignas(16) bf16 Ax[64 * 40];      // 64 pts x K32, ld 40
  const int p0 = blockIdx.x * 64;
  for (int idx = threadIdx.x; idx < 2048; idx += 128) {
    int m = idx >> 5, k = idx & 31;
    Ax[m * 40 + k] = (k < 12) ? tobf(x[(p0 + m) * 12 + k]) : tobf(0.f);
  }
  __syncthreads();
  const bf16* W = ws + WS_INW;
  const int wave = threadIdx.x >> 5, lane = threadIdx.x & 31;
  const int nlo = lane & 15, mofs = (lane >> 4) << 3;
  for (int j = 0; j < 4; ++j) {
    int t = wave * 4 + j, mt = t >> 2, nt = t & 3;
    int o = nt * 16 + nlo;
    v8f acc = bias_acc(in_b[o]);
    acc = wmma_bf16(load_a16(Ax + mt * 16 * 40, 40),
                    load_bt16(W + nt * 16 * 32, 32), acc);
    for (int r = 0; r < 8; ++r) {
      int m = mt * 16 + r + mofs;
      h[(p0 + m) * 64 + o] = acc[r];
    }
  }
}

// ---------------- forward DFT per slab: F = Wdft @ slab ---------------------
// dir 0: transform along n (slab = (b,row m)); dir 1: along m (slab = (b,col n))
__global__ void fno_dft(const float* __restrict__ h, bf16* __restrict__ ws,
                        int dir) {
  __shared__ alignas(16) bf16 At[64 * 264];     // slab^T: 64ch x 256, ld 264
  const int s = blockIdx.x;
  int base, step;
  if (dir == 0) { base = s * 16384; step = 64; }
  else          { base = (s >> 8) * 4194304 + (s & 255) * 64; step = 16384; }
  for (int idx = threadIdx.x; idx < 16384; idx += 256) {
    int n = idx >> 6, c = idx & 63;
    At[c * 264 + n] = tobf(h[base + n * step + c]);
  }
  __syncthreads();
  const bf16* A = ws + WS_DFT;
  bf16* Fr = ws + WS_FR;
  bf16* Fi = ws + WS_FI;
  const int wave = threadIdx.x >> 5, lane = threadIdx.x & 31;
  const int nlo = lane & 15, mofs = (lane >> 4) << 3;
  for (int j = 0; j < 2; ++j) {
    int t = wave * 2 + j, mt = t >> 2, nt = t & 3;  // M: 64 mode-rows, N: 64 ch
    v8f acc = {};
    for (int kk = 0; kk < 8; ++kk)
      acc = wmma_bf16(load_a16(A + mt * 16 * 256 + kk * 32, 256),
                      load_bt16(At + nt * 16 * 264 + kk * 32, 264), acc);
    bf16* F = (mt < 2) ? Fr : Fi;
    int rb = mt * 16 - ((mt < 2) ? 0 : 32);
    int c = nt * 16 + nlo;
    for (int r = 0; r < 8; ++r) {
      int mode = rb + r + mofs;                     // 0..31
      F[(long)mode * 131072 + s * 64 + c] = tobf(acc[r]);
    }
  }
}

// ---------------- per-mode complex channel mixing ---------------------------
// Gr = Fr*Wr - Fi*Wi ; Gi = Fr*Wi + Fi*Wr ; batched over 2048 slabs per mode.
// Output layout [kcat][slab][o] keeps D-fragment stores 32B-contiguous.
__global__ void fno_mix(bf16* __restrict__ ws, int l, int dir) {
  const int stile = blockIdx.x;                     // 64-slab tiles: 0..31
  const int k = blockIdx.y;                         // mode 0..31
  const bf16* Fr = ws + WS_FR + (long)k * 131072 + stile * 4096;
  const bf16* Fi = ws + WS_FI + (long)k * 131072 + stile * 4096;
  const long wsel = ((long)((l * 2 + dir) * 32 + k)) * 4096;
  const bf16* Wr = ws + WS_MWR + wsel;
  const bf16* Wi = ws + WS_MWI + wsel;
  const bf16* Wn = ws + WS_MWN + wsel;
  bf16* G = ws + WS_GT;
  const int wave = threadIdx.x >> 5, lane = threadIdx.x & 31;
  const int nlo = lane & 15, mofs = (lane >> 4) << 3;
  for (int j = 0; j < 8; ++j) {
    int t = wave * 8 + j;                           // 0..31
    int isIm = t >> 4;
    int tt = t & 15, mt = tt >> 2, nt = tt & 3;
    v8f acc = {};
    for (int kk = 0; kk < 2; ++kk) {
      const bf16* b1 = (isIm ? Wi : Wr) + nt * 16 * 64 + kk * 32;
      const bf16* b2 = (isIm ? Wr : Wn) + nt * 16 * 64 + kk * 32;
      acc = wmma_bf16(load_a16(Fr + mt * 1024 + kk * 32, 64), load_bt16(b1, 64), acc);
      acc = wmma_bf16(load_a16(Fi + mt * 1024 + kk * 32, 64), load_bt16(b2, 64), acc);
    }
    int kcat = isIm ? (32 + k) : k;
    int o = nt * 16 + nlo;
    bf16* Gk = G + (long)kcat * 131072;
    for (int r = 0; r < 8; ++r) {
      int srow = stile * 64 + mt * 16 + r + mofs;
      Gk[srow * 64 + o] = tobf(acc[r]);
    }
  }
}

// ---------------- inverse DFT per slab: out = V @ [Gr;Gi] -------------------
__global__ void fno_idft(const bf16* __restrict__ ws, float* __restrict__ sbuf,
                         int dir) {
  __shared__ alignas(16) bf16 Bt[64 * 72];        // G[s]^T: 64o x 64kcat, ld 72
  const int s = blockIdx.x;
  // transpose during staging: global reads coalesced (o contiguous)
  for (int idx = threadIdx.x; idx < 4096; idx += 256) {
    int kcat = idx >> 6, o = idx & 63;
    Bt[o * 72 + kcat] = ws[WS_GT + (long)kcat * 131072 + s * 64 + o];
  }
  __syncthreads();
  const bf16* V = ws + WS_IDFT;
  int base, step;
  if (dir == 0) { base = s * 16384; step = 64; }
  else          { base = (s >> 8) * 4194304 + (s & 255) * 64; step = 16384; }
  const int wave = threadIdx.x >> 5, lane = threadIdx.x & 31;
  const int nlo = lane & 15, mofs = (lane >> 4) << 3;
  for (int j = 0; j < 8; ++j) {
    int t = wave * 8 + j, mt = t >> 2, nt = t & 3;  // M: 256 positions, N: 64 ch
    v8f acc = {};
    for (int kk = 0; kk < 2; ++kk)
      acc = wmma_bf16(load_a16(V + mt * 16 * 64 + kk * 32, 64),
                      load_bt16(Bt + nt * 16 * 72 + kk * 32, 72), acc);
    int o = nt * 16 + nlo;
    for (int r = 0; r < 8; ++r) {
      int row = mt * 16 + r + mofs;
      float* p = sbuf + base + row * step + o;
      if (dir == 0) *p = acc[r];
      else          *p += acc[r];
    }
  }
}

// ---------------- channel MLP + residual: h += W2 relu(W1 s + b1) + b2 ------
__global__ void fno_mlp(const float* __restrict__ sbuf, const bf16* __restrict__ ws,
                        const float* __restrict__ bc_b1, const float* __restrict__ bc_b2,
                        float* __restrict__ h, int l) {
  __shared__ alignas(16) bf16 sA[64 * 72];        // 64 pts x 64 ch, ld 72
  __shared__ alignas(16) bf16 hA[64 * 136];       // 64 pts x 128 hid, ld 136
  const int p0 = blockIdx.x * 64;
  for (int idx = threadIdx.x; idx < 4096; idx += 128) {
    int m = idx >> 6, c = idx & 63;
    sA[m * 72 + c] = tobf(sbuf[(long)p0 * 64 + idx]);
  }
  __syncthreads();
  const bf16* W1 = ws + WS_BCW1 + l * 8192;
  const bf16* W2 = ws + WS_BCW2 + l * 8192;
  const float* b1 = bc_b1 + l * 128;
  const float* b2 = bc_b2 + l * 64;
  const int wave = threadIdx.x >> 5, lane = threadIdx.x & 31;
  const int nlo = lane & 15, mofs = (lane >> 4) << 3;
  for (int j = 0; j < 8; ++j) {                     // GEMM1: 64x128, K=64
    int t = wave * 8 + j, mt = t & 3, nt = t >> 2;
    int o = nt * 16 + nlo;
    v8f acc = bias_acc(b1[o]);
    for (int kk = 0; kk < 2; ++kk)
      acc = wmma_bf16(load_a16(sA + mt * 16 * 72 + kk * 32, 72),
                      load_bt16(W1 + nt * 16 * 64 + kk * 32, 64), acc);
    for (int r = 0; r < 8; ++r) {
      float v = acc[r];
      hA[(mt * 16 + r + mofs) * 136 + o] = tobf(v > 0.f ? v : 0.f);
    }
  }
  __syncthreads();
  for (int j = 0; j < 4; ++j) {                     // GEMM2: 64x64, K=128
    int t = wave * 4 + j, mt = t >> 2, nt = t & 3;
    int o = nt * 16 + nlo;
    v8f acc = bias_acc(b2[o]);
    for (int kk = 0; kk < 4; ++kk)
      acc = wmma_bf16(load_a16(hA + mt * 16 * 136 + kk * 32, 136),
                      load_bt16(W2 + nt * 16 * 128 + kk * 32, 128), acc);
    for (int r = 0; r < 8; ++r) {
      long gi = (long)(p0 + mt * 16 + r + mofs) * 64 + o;
      h[gi] = h[gi] + acc[r];
    }
  }
}

// ---------------- output heads ----------------------------------------------
__global__ void fno_heads(const float* __restrict__ h, const bf16* __restrict__ ws,
                          const float* __restrict__ ob1, const float* __restrict__ ob2,
                          const float* __restrict__ rb1, const float* __restrict__ rb2,
                          float* __restrict__ fout, float* __restrict__ rout) {
  __shared__ alignas(16) bf16 hA[64 * 72];
  __shared__ alignas(16) bf16 tA[64 * 136];
  const int p0 = blockIdx.x * 64;
  for (int idx = threadIdx.x; idx < 4096; idx += 128) {
    int m = idx >> 6, c = idx & 63;
    hA[m * 72 + c] = tobf(h[(long)p0 * 64 + idx]);
  }
  __syncthreads();
  const int wave = threadIdx.x >> 5, lane = threadIdx.x & 31;
  const int nlo = lane & 15, mofs = (lane >> 4) << 3;
  // forecast stage 1: t = h @ out_w1^T + ob1
  const bf16* OW1 = ws + WS_OW1;
  for (int j = 0; j < 8; ++j) {
    int t = wave * 8 + j, mt = t & 3, nt = t >> 2;
    int o = nt * 16 + nlo;
    v8f acc = bias_acc(ob1[o]);
    for (int kk = 0; kk < 2; ++kk)
      acc = wmma_bf16(load_a16(hA + mt * 16 * 72 + kk * 32, 72),
                      load_bt16(OW1 + nt * 16 * 64 + kk * 32, 64), acc);
    for (int r = 0; r < 8; ++r)
      tA[(mt * 16 + r + mofs) * 136 + o] = tobf(acc[r]);
  }
  __syncthreads();
  // forecast stage 2: 5 (padded 16) outputs
  const bf16* OW2 = ws + WS_OW2;
  {
    int mt = wave;
    v8f acc = {};
    for (int kk = 0; kk < 4; ++kk)
      acc = wmma_bf16(load_a16(tA + mt * 16 * 136 + kk * 32, 136),
                      load_bt16(OW2 + kk * 32, 128), acc);
    int o = nlo;
    if (o < 5) {
      float bias = ob2[o];
      for (int r = 0; r < 8; ++r)
        fout[(long)(p0 + mt * 16 + r + mofs) * 5 + o] = acc[r] + bias;
    }
  }
  __syncthreads();
  // residual head stage 1
  const bf16* RW1 = ws + WS_RW1;
  for (int j = 0; j < 8; ++j) {
    int t = wave * 8 + j, mt = t & 3, nt = t >> 2;
    int o = nt * 16 + nlo;
    v8f acc = bias_acc(rb1[o]);
    for (int kk = 0; kk < 2; ++kk)
      acc = wmma_bf16(load_a16(hA + mt * 16 * 72 + kk * 32, 72),
                      load_bt16(RW1 + nt * 16 * 64 + kk * 32, 64), acc);
    for (int r = 0; r < 8; ++r)
      tA[(mt * 16 + r + mofs) * 136 + o] = tobf(acc[r]);
  }
  __syncthreads();
  // residual head stage 2
  const bf16* RW2 = ws + WS_RW2;
  for (int j = 0; j < 4; ++j) {
    int t = wave * 4 + j, mt = t >> 2, nt = t & 3;
    int o = nt * 16 + nlo;
    v8f acc = bias_acc(rb2[o]);
    for (int kk = 0; kk < 4; ++kk)
      acc = wmma_bf16(load_a16(tA + mt * 16 * 136 + kk * 32, 136),
                      load_bt16(RW2 + nt * 16 * 128 + kk * 32, 128), acc);
    for (int r = 0; r < 8; ++r)
      rout[(long)(p0 + mt * 16 + r + mofs) * 64 + o] = acc[r];
  }
}

// ---------------- launcher ---------------------------------------------------
extern "C" void kernel_launch(void* const* d_in, const int* in_sizes, int n_in,
                              void* d_out, int out_size, void* d_ws, size_t ws_size,
                              hipStream_t stream) {
  (void)in_sizes; (void)n_in; (void)out_size; (void)ws_size;
  const float* x     = (const float*)d_in[0];
  const float* in_w  = (const float*)d_in[1];
  const float* in_b  = (const float*)d_in[2];
  const float* fw    = (const float*)d_in[3];
  const float* bc_w1 = (const float*)d_in[4];
  const float* bc_b1 = (const float*)d_in[5];
  const float* bc_w2 = (const float*)d_in[6];
  const float* bc_b2 = (const float*)d_in[7];
  const float* ow1   = (const float*)d_in[8];
  const float* ob1   = (const float*)d_in[9];
  const float* ow2   = (const float*)d_in[10];
  const float* ob2   = (const float*)d_in[11];
  const float* rw1   = (const float*)d_in[12];
  const float* rb1   = (const float*)d_in[13];
  const float* rw2   = (const float*)d_in[14];
  const float* rb2   = (const float*)d_in[15];

  float* fout = (float*)d_out;                 // forecast: 8*256*256*1*5
  float* rout = fout + 2621440;                // res_out:  8*256*256*64
  float* hbuf = rout + 33554432;               // h:        8*256*256*64
  float* sbuf = rout;                          // reuse res segment as s scratch
  bf16*  ws   = (bf16*)d_ws;

  fno_prep<<<2048, 256, 0, stream>>>(in_w, bc_w1, bc_w2, ow1, ow2, rw1, rw2, fw, ws);
  fno_in<<<8192, 128, 0, stream>>>(x, ws, in_b, hbuf);
  for (int l = 0; l < 4; ++l) {
    for (int dir = 0; dir < 2; ++dir) {
      fno_dft<<<2048, 256, 0, stream>>>(hbuf, ws, dir);
      fno_mix<<<dim3(32, 32), 128, 0, stream>>>(ws, l, dir);
      fno_idft<<<2048, 256, 0, stream>>>(ws, sbuf, dir);
    }
    fno_mlp<<<8192, 128, 0, stream>>>(sbuf, ws, bc_b1, bc_b2, hbuf, l);
  }
  fno_heads<<<8192, 128, 0, stream>>>(hbuf, ws, ob1, ob2, rb1, rb2, fout, rout);
}
